// PathEmbedding_32607391711718
// MI455X (gfx1250) — compile-verified, matrix operand order checked by reading, not run on hardware
//
#include <hip/hip_runtime.h>
#include <hip/hip_bf16.h>
#include <math.h>

typedef __attribute__((ext_vector_type(16))) _Float16 v16h;
typedef __attribute__((ext_vector_type(8)))  float    v8f;

// Problem constants (match reference)
constexpr int Bc   = 4;
constexpr int Nc   = 10000;
constexpr int Hc   = 256;
constexpr int Rc   = 8;
constexpr int EMBc = 32;
constexpr int Lc   = 200000;
constexpr int Pc   = 20000;

__device__ __forceinline__ float4 f4add3(float4 a, float4 b, float4 c) {
  return make_float4(a.x + b.x + c.x, a.y + b.y + c.y, a.z + b.z + c.z, a.w + b.w + c.w);
}

// ---------------------------------------------------------------------------
// K0: build segment start offsets from sorted path_index (CSR-style)
// ---------------------------------------------------------------------------
__global__ void k_seg_start(const int* __restrict__ path_index, int* __restrict__ seg_start) {
  int l = blockIdx.x * blockDim.x + threadIdx.x;
  if (l >= Lc) return;
  int pi   = path_index[l];
  int prev = (l == 0) ? -1 : path_index[l - 1];
  for (int p = prev + 1; p <= pi; ++p) seg_start[p] = l;
  if (l == Lc - 1) {
    for (int p = pi + 1; p <= Pc; ++p) seg_start[p] = Lc;
  }
}

// ---------------------------------------------------------------------------
// K0b: zero the global accumulator (ws is poisoned, not zeroed)
// ---------------------------------------------------------------------------
__global__ void k_zero(float* __restrict__ gacc) {
  int i = blockIdx.x * blockDim.x + threadIdx.x;
  if (i < Bc * Hc) gacc[i] = 0.0f;
}

// ---------------------------------------------------------------------------
// K1: LayerNorm(x) @ Wp + bp -> xp (B,N,EMB) via v_wmma_f32_16x16x32_f16.
//  - 2 waves/block, one 16-row M-tile per wave (10000 % 16 == 0: no straddle).
//  - Wp staged once per block into LDS as f16, pre-swizzled into the exact
//    WMMA B-register layout -> each v16h operand is one contiguous LDS load.
//  - The wave's 16x256 f32 x-tile is staged into LDS with
//    GLOBAL_LOAD_ASYNC_TO_LDS_B128 (ASYNCcnt), then read twice (stats + A).
// LDS layout (single buffer, base offset 0):
//   [0, 32KB)  : x tiles, 16KB per wave slot
//   [32KB,48KB): Wp f16, index ((kk*2+t)*32 + lane)*16 + i
// ---------------------------------------------------------------------------
constexpr int K1_WAVES   = 2;
constexpr int K1_XBYTES  = K1_WAVES * 16 * Hc * 4;        // 32 KB
constexpr int K1_WPHALF  = 8 * 2 * 32 * 16;               // 8192 f16 = 16 KB

__global__ void k_ln_proj(const float* __restrict__ x,
                          const float* __restrict__ ln_g, const float* __restrict__ ln_b,
                          const float* __restrict__ Wp,   const float* __restrict__ bp,
                          float* __restrict__ xp) {
  __shared__ __align__(16) unsigned char smem[K1_XBYTES + K1_WPHALF * 2];
  float*    xt  = (float*)smem;
  _Float16* wpB = (_Float16*)(smem + K1_XBYTES);

  // -- cooperative Wp stage into WMMA B layout --------------------------------
  for (int idx = threadIdx.x; idx < K1_WPHALF; idx += blockDim.x) {
    int i  = idx & 15;
    int ln = (idx >> 4) & 31;
    int t  = (idx >> 9) & 1;
    int kk = idx >> 10;
    int k  = ((ln >> 4) << 3) + (i & 7) + ((i >> 3) << 4);
    wpB[idx] = (_Float16)Wp[(kk * 32 + k) * EMBc + t * 16 + (ln & 15)];
  }
  __syncthreads();

  int wslot = threadIdx.x >> 5;
  int lane  = threadIdx.x & 31;
  int wave  = blockIdx.x * K1_WAVES + wslot;      // grid sized exactly: no guard
  int m     = lane & 15;
  int h4    = lane >> 4;
  int koff  = h4 * 8;
  int rowBase = wave * 16;

  // -- async stage of this wave's 16x256 f32 tile into LDS --------------------
  unsigned ldsbase = (unsigned)(wslot * 16 * Hc * 4);
  uint64_t gbase   = (uint64_t)(uintptr_t)x + (uint64_t)rowBase * Hc * 4;
  for (int it = 0; it < 32; ++it) {               // 32 issues x 512B = 16KB
    unsigned loff = ldsbase + it * 512 + lane * 16;
    uint64_t ga   = gbase   + it * 512 + lane * 16;
    asm volatile("global_load_async_to_lds_b128 %0, %1, off"
                 :: "v"(loff), "v"(ga) : "memory");
  }
  asm volatile("s_wait_asynccnt 0x0" ::: "memory");

  const float* xrow = xt + wslot * 16 * Hc + m * Hc;

  // -- mean/var over H=256 (each lane of the pair sums 128 elems) -------------
  float sum = 0.0f, sumsq = 0.0f;
  const float4* xr4 = (const float4*)(xrow + h4 * 128);
  #pragma unroll 4
  for (int j = 0; j < 32; ++j) {
    float4 v = xr4[j];
    sum   += v.x + v.y + v.z + v.w;
    sumsq += v.x * v.x + v.y * v.y + v.z * v.z + v.w * v.w;
  }
  sum   += __shfl_xor(sum,   16, 32);
  sumsq += __shfl_xor(sumsq, 16, 32);
  float mu  = sum * (1.0f / 256.0f);
  float var = sumsq * (1.0f / 256.0f) - mu * mu;
  float rs  = rsqrtf(var + 1e-5f);

  // -- K loop: 8 steps of K=32 ------------------------------------------------
  v8f c0 = {}; v8f c1 = {};
  int n = lane & 15;
  for (int kk = 0; kk < 8; ++kk) {
    int kb = kk * 32;
    v16h a;
    #pragma unroll
    for (int i = 0; i < 16; ++i) {
      int hcol = kb + koff + (i & 7) + ((i >> 3) << 4);
      float v  = (xrow[hcol] - mu) * rs * ln_g[hcol] + ln_b[hcol];
      a[i] = (_Float16)v;
    }
    v16h b0 = *(const v16h*)(wpB + ((kk * 2 + 0) * 32 + lane) * 16);
    v16h b1 = *(const v16h*)(wpB + ((kk * 2 + 1) * 32 + lane) * 16);
    c0 = __builtin_amdgcn_wmma_f32_16x16x32_f16(false, a, false, b0, (short)0, c0, false, false);
    c1 = __builtin_amdgcn_wmma_f32_16x16x32_f16(false, a, false, b1, (short)0, c1, false, false);
  }
  // C layout: elem j -> row (j + 8*h4), col (lane&15)
  #pragma unroll
  for (int j = 0; j < 8; ++j) {
    float* dst = xp + (size_t)(rowBase + j + 8 * h4) * EMBc;
    dst[n]      = c0[j] + bp[n];
    dst[16 + n] = c1[j] + bp[16 + n];
  }
}

// ---------------------------------------------------------------------------
// K2: feat = xp[b,path_list] + edge + pos ; s = tanh(feat@W1+b1)@W2+b2
// One wave per (b, 16-path tile). feat@W1: single v_wmma_f32_16x16x32_f16
// (K=32 exact, N padded 8->16 with zeros). t@W2 (8x8) via intra-wave LDS.
// A operand assembled from float4 loads (contiguous 8-element K runs).
// ---------------------------------------------------------------------------
__global__ void k_feat_s(const float* __restrict__ xp,
                         const float* __restrict__ edge_table, const float* __restrict__ pos_table,
                         const float* __restrict__ W1, const float* __restrict__ b1,
                         const float* __restrict__ W2, const float* __restrict__ b2,
                         const int* __restrict__ path_list, const int* __restrict__ path_edge_type,
                         const int* __restrict__ path_positions,
                         float* __restrict__ s_out, int numTiles) {
  __shared__ float lds_t[8][16][8];             // [wave][path row][r]
  int wslot = threadIdx.x >> 5;
  int wave  = blockIdx.x * (blockDim.x >> 5) + wslot;
  if (wave >= numTiles) return;
  int lane = threadIdx.x & 31;
  int m = lane & 15, h4 = lane >> 4, koff = h4 * 8, n = lane & 15;

  int bIdx  = wave / (Lc / 16);
  int lbase = (wave % (Lc / 16)) * 16;
  int l  = lbase + m;
  int node = path_list[l];
  int et   = path_edge_type[l];
  int pp   = path_positions[l];
  const float4* xp4 = (const float4*)(xp + ((size_t)bIdx * Nc + node) * EMBc);
  const float4* e4  = (const float4*)(edge_table + (size_t)et * EMBc);
  const float4* p4  = (const float4*)(pos_table  + (size_t)pp * EMBc);

  // K runs for this lane: e in [koff,koff+8) and [koff+16,koff+24)
  int q = 2 * h4;
  float4 A0 = f4add3(xp4[q],     e4[q],     p4[q]);
  float4 A1 = f4add3(xp4[q + 1], e4[q + 1], p4[q + 1]);
  float4 A2 = f4add3(xp4[q + 4], e4[q + 4], p4[q + 4]);
  float4 A3 = f4add3(xp4[q + 5], e4[q + 5], p4[q + 5]);
  float fr[16] = {A0.x, A0.y, A0.z, A0.w, A1.x, A1.y, A1.z, A1.w,
                  A2.x, A2.y, A2.z, A2.w, A3.x, A3.y, A3.z, A3.w};
  v16h a, bm;
  #pragma unroll
  for (int i = 0; i < 16; ++i) {
    int k = koff + (i & 7) + ((i >> 3) << 4);
    a[i]  = (_Float16)fr[i];
    bm[i] = (n < Rc) ? (_Float16)W1[k * Rc + n] : (_Float16)0.0f;
  }
  v8f c = {};
  c = __builtin_amdgcn_wmma_f32_16x16x32_f16(false, a, false, bm, (short)0, c, false, false);

  // tanh + redistribute through LDS (DS ops from one wave are in-order)
  if (n < Rc) {
    #pragma unroll
    for (int j = 0; j < 8; ++j)
      lds_t[wslot][j + 8 * h4][n] = tanhf(c[j] + b1[n]);
  }
  asm volatile("s_wait_dscnt 0" ::: "memory");

  // s[m2][r2] = b2[r2] + sum_r t[m2][r]*W2[r][r2]; lane does m2=n, r2 in [4*h4,4*h4+4)
  float* srow = s_out + ((size_t)bIdx * Lc + (lbase + n)) * Rc;
  #pragma unroll
  for (int qq = 0; qq < 4; ++qq) {
    int r2 = h4 * 4 + qq;
    float acc = b2[r2];
    #pragma unroll
    for (int r = 0; r < Rc; ++r) acc += lds_t[wslot][n][r] * W2[r * Rc + r2];
    srow[r2] = acc;
  }
}

// ---------------------------------------------------------------------------
// K3: per-segment softmax + weighted outer-product accumulation.
// One wave per segment (strided); lane = (b<<3)|r. feat recomputed from xp.
// Cooperative float4 load + __shfl broadcast inside each 8-lane b-group.
// ---------------------------------------------------------------------------
__global__ void k_segments(const float* __restrict__ xp,
                           const float* __restrict__ edge_table, const float* __restrict__ pos_table,
                           const float* __restrict__ s_arr, const float* __restrict__ path_weight,
                           const int* __restrict__ path_list, const int* __restrict__ path_edge_type,
                           const int* __restrict__ path_positions,
                           const int* __restrict__ seg_start, float* __restrict__ gacc) {
  __shared__ float blockAcc[Bc * Hc];           // 1024 floats
  for (int i = threadIdx.x; i < Bc * Hc; i += blockDim.x) blockAcc[i] = 0.0f;
  __syncthreads();

  int lane  = threadIdx.x & 31;
  int gw    = blockIdx.x * (blockDim.x >> 5) + (threadIdx.x >> 5);
  int nW    = gridDim.x * (blockDim.x >> 5);
  int bIdx  = lane >> 3;
  int r     = lane & 7;
  const float* sb = s_arr + (size_t)bIdx * Lc * Rc + r;

  float acc[EMBc];
  #pragma unroll
  for (int e = 0; e < EMBc; ++e) acc[e] = 0.0f;

  for (int p = gw; p < Pc; p += nW) {
    int st = seg_start[p], en = seg_start[p + 1];
    if (st >= en) continue;                     // empty segment: hp[p] = 0
    float mx = -3.4e38f;
    for (int l = st; l < en; ++l) {
      __builtin_prefetch(sb + (size_t)(l + 4) * Rc, 0, 0);   // global_prefetch_b8
      mx = fmaxf(mx, sb[(size_t)l * Rc]);
    }
    float dn = 0.0f;
    for (int l = st; l < en; ++l) dn += __expf(sb[(size_t)l * Rc] - mx);
    float pw = path_weight[p] / dn;
    for (int l = st; l < en; ++l) {
      float aw = __expf(sb[(size_t)l * Rc] - mx) * pw;       // pw * attn[l,b,r]
      int node = path_list[l], et = path_edge_type[l], pp = path_positions[l];
      const float4* xpr = (const float4*)(xp + ((size_t)bIdx * Nc + node) * EMBc);
      const float4* er  = (const float4*)(edge_table + (size_t)et * EMBc);
      const float4* pr  = (const float4*)(pos_table  + (size_t)pp * EMBc);
      float4 f4 = f4add3(xpr[r], er[r], pr[r]);
      #pragma unroll
      for (int e = 0; e < EMBc; ++e) {
        int src = (bIdx << 3) + (e >> 2);
        float comp = ((e & 3) == 0) ? f4.x : ((e & 3) == 1) ? f4.y
                   : ((e & 3) == 2) ? f4.z : f4.w;
        float fe = __shfl(comp, src, 32);       // ds_bpermute broadcast
        acc[e] += aw * fe;
      }
    }
  }
  // merge: registers -> LDS atomics -> global atomics (once per block)
  #pragma unroll
  for (int e = 0; e < EMBc; ++e)
    atomicAdd(&blockAcc[(bIdx * Rc + r) * EMBc + e], acc[e]);
  __syncthreads();
  for (int i = threadIdx.x; i < Bc * Hc; i += blockDim.x)
    atomicAdd(&gacc[i], blockAcc[i]);
}

// ---------------------------------------------------------------------------
// K4: out = acc / P
// ---------------------------------------------------------------------------
__global__ void k_final(const float* __restrict__ gacc, float* __restrict__ out) {
  int i = blockIdx.x * blockDim.x + threadIdx.x;
  if (i < Bc * Hc) out[i] = gacc[i] * (1.0f / (float)Pc);
}

// ---------------------------------------------------------------------------
extern "C" void kernel_launch(void* const* d_in, const int* in_sizes, int n_in,
                              void* d_out, int out_size, void* d_ws, size_t ws_size,
                              hipStream_t stream) {
  const float* x           = (const float*)d_in[0];
  const float* path_weight = (const float*)d_in[1];
  const float* ln_g        = (const float*)d_in[2];
  const float* ln_b        = (const float*)d_in[3];
  const float* edge_table  = (const float*)d_in[4];
  const float* pos_table   = (const float*)d_in[5];
  const float* Wp          = (const float*)d_in[6];
  const float* bp          = (const float*)d_in[7];
  const float* W1          = (const float*)d_in[8];
  const float* b1          = (const float*)d_in[9];
  const float* W2          = (const float*)d_in[10];
  const float* b2          = (const float*)d_in[11];
  const int*   path_list      = (const int*)d_in[12];
  const int*   path_index     = (const int*)d_in[13];
  const int*   path_edge_type = (const int*)d_in[14];
  const int*   path_positions = (const int*)d_in[15];
  float* out = (float*)d_out;

  // workspace carve-up (~31 MB)
  float* xp    = (float*)d_ws;                       // B*N*EMB   = 1,280,000 f32
  float* s_arr = xp + (size_t)Bc * Nc * EMBc;        // B*L*R     = 6,400,000 f32
  float* gacc  = s_arr + (size_t)Bc * Lc * Rc;       // B*H       = 1,024 f32
  int*   seg_start = (int*)(gacc + Bc * Hc);         // P+1 ints

  k_seg_start<<<(Lc + 255) / 256, 256, 0, stream>>>(path_index, seg_start);
  k_zero<<<(Bc * Hc + 255) / 256, 256, 0, stream>>>(gacc);

  int tiles1 = (Bc * Nc) / 16;                       // 2500 (divisible by K1_WAVES)
  k_ln_proj<<<tiles1 / K1_WAVES, K1_WAVES * 32, 0, stream>>>(x, ln_g, ln_b, Wp, bp, xp);

  int tiles2 = Bc * (Lc / 16);                       // 50000
  k_feat_s<<<(tiles2 + 7) / 8, 256, 0, stream>>>(xp, edge_table, pos_table, W1, b1, W2, b2,
                                                 path_list, path_edge_type, path_positions,
                                                 s_arr, tiles2);

  k_segments<<<256, 256, 0, stream>>>(xp, edge_table, pos_table, s_arr, path_weight,
                                      path_list, path_edge_type, path_positions,
                                      seg_start, gacc);

  k_final<<<(Bc * Hc + 255) / 256, 256, 0, stream>>>(gacc, out);
}
